// CFTAuxHead_82884278879205
// MI455X (gfx1250) — matrix operand order repaired and checked by reading, not compiled, not run on hardware
//
#include <hip/hip_runtime.h>
#include <hip/hip_bf16.h>

// CDNA5 (gfx1250) wave32 WMMA types
typedef __attribute__((ext_vector_type(16))) _Float16 v16h;
typedef __attribute__((ext_vector_type(8)))  float    v8f;
typedef __attribute__((ext_vector_type(4)))  int      v4i;

#define HIN    160
#define HOUT   640
#define NBOX   128
#define NBATCH 32
#define STRIPS 40                      // 640 rows / 16-row strip
#define NBLK   (NBATCH * STRIPS)       // 1280 partials

#if __has_builtin(__builtin_amdgcn_global_load_async_to_lds_b128) && \
    __has_builtin(__builtin_amdgcn_s_wait_asynccnt)
#define USE_ASYNC_LDS 1
typedef __attribute__((address_space(1))) v4i gv4i;   // global int4
typedef __attribute__((address_space(3))) v4i lv4i;   // LDS int4
#endif

// frac table for src = (p+0.5)/4 - 0.5 ; p mod 4 -> frac (all exact in f16)
__device__ __forceinline__ float frac4(int m) {
    return (m == 0) ? 0.625f : (m == 1) ? 0.875f : (m == 2) ? 0.125f : 0.375f;
}

__global__ __launch_bounds__(256) void cft_main(const float* __restrict__ feat,
                                                const float* __restrict__ boxes,
                                                float* __restrict__ partial)
{
    __shared__ _Float16 T[16][192];    // row-interp strip (f16, zero-pad cols>=160)
    __shared__ _Float16 Bm[HOUT][32];  // per-out-col column weights (K window)
    __shared__ float    Fs[6][HIN];    // staged input rows (async copy target)
    __shared__ float    Hs[16][HOUT];  // rasterized heatmap strip
    __shared__ int4     rectS[NBOX];   // {xmin,xmax,ymin,ymax}
    __shared__ float    zS[NBOX];
    __shared__ float    redS[256];

    const int tid  = threadIdx.x;
    const int lane = tid & 31;
    const int wv   = tid >> 5;          // 8 waves
    const int b    = blockIdx.y;
    const int r0   = blockIdx.x * 16;   // first output row of strip
    const int ir0  = 4 * (int)blockIdx.x - 1;   // first (unclamped) input row

    const float* fb = feat + (size_t)b * HIN * HIN;

    // --- stage the 6 needed input rows into LDS ------------------------
#ifdef USE_ASYNC_LDS
    if (tid < 240) {                    // 6 rows * 160 f32 = 240 x 16B
        int slot = tid / 40;
        int c4   = (tid - slot * 40) * 4;
        int ir   = ir0 + slot;
        ir = ir < 0 ? 0 : (ir > HIN - 1 ? HIN - 1 : ir);
        const float* src = fb + (size_t)ir * HIN + c4;
        __builtin_amdgcn_global_load_async_to_lds_b128(
            (gv4i*)src, (lv4i*)&Fs[slot][c4], 0, 0);
    }
#else
    for (int p = tid; p < 6 * HIN; p += 256) {
        int slot = p / HIN, ic = p - slot * HIN;
        int ir = ir0 + slot;
        ir = ir < 0 ? 0 : (ir > HIN - 1 ? HIN - 1 : ir);
        Fs[slot][ic] = fb[(size_t)ir * HIN + ic];
    }
#endif

    // --- build Bm: column-interp weights (independent of staged data) ---
    for (int c = tid; c < HOUT; c += 256) {
        int j  = c >> 4;
        int kc = j ? (4 * j - 2) : 0;           // even K-window base
        _Float16* col = &Bm[c][0];
        #pragma unroll
        for (int i = 0; i < 32; ++i) col[i] = (_Float16)0.f;
        int q  = c >> 2, m = c & 3;
        int i0 = q + ((m < 2) ? -1 : 0);
        float fr = frac4(m);
        int iA = (i0 < 0) ? 0 : i0;
        int iB = (i0 + 1 > HIN - 1) ? (HIN - 1) : (i0 + 1);
        col[iA - kc] += (_Float16)(1.f - fr);   // += handles edge-clamped tap
        col[iB - kc] += (_Float16)fr;
    }

    // --- box rects (reference int/trunc/clamp semantics) ----------------
    if (tid < NBOX) {
        const float* bx = boxes + ((size_t)b * NBOX + tid) * 5;
        float x = bx[0], y = bx[1], z = bx[2], width = bx[3], length = bx[4];
        bool valid = (width > 0.f) && (length > 0.f);
        float w  = fmaxf(floorf(width * 0.5f), 3.0f);
        float h  = fmaxf(floorf(length * 0.5f), 3.0f);
        float cx = (float)(int)x, cy = (float)(int)y;
        int xmin = max((int)(cx - w), 0);
        int xmax = min((int)(cx + w + 1.f), HOUT);
        int ymin = max((int)(cy - h), 0);
        int ymax = min((int)(cy + h + 1.f), HOUT);
        if (!valid) { xmin = 0; xmax = 0; }
        rectS[tid] = make_int4(xmin, xmax, ymin, ymax);
        zS[tid] = z;
    }

#ifdef USE_ASYNC_LDS
    __builtin_amdgcn_s_wait_asynccnt(0);
#endif
    __syncthreads();                    // Fs + Bm + rects visible

    // --- build T from staged rows (row interpolation, f16) --------------
    for (int p = tid; p < 16 * 192; p += 256) {
        int rr = p / 192, ic = p - rr * 192;
        float v = 0.f;
        if (ic < HIN) {
            int r  = r0 + rr;
            int q  = r >> 2, m = r & 3;
            int i0 = q + ((m < 2) ? -1 : 0);
            float fr = frac4(m);
            int iA = (i0 < 0) ? 0 : i0;
            int iB = (i0 + 1 > HIN - 1) ? (HIN - 1) : (i0 + 1);
            v = Fs[iA - ir0][ic] * (1.f - fr) + Fs[iB - ir0][ic] * fr;
        }
        T[rr][ic] = (_Float16)v;
    }

    // --- rasterize heatmap strip: wave wv owns rows 2wv, 2wv+1 ----------
    // LDS ops from one wave are in-order => sequential box overwrite
    // semantics ("last box wins") are preserved exactly.
    for (int rr = 2 * wv; rr <= 2 * wv + 1; ++rr) {
        for (int c = lane; c < HOUT; c += 32) Hs[rr][c] = 0.f;   // own-row zero
        int r = r0 + rr;
        for (int jb = 0; jb < NBOX; ++jb) {
            int4 rc = rectS[jb];                 // one ds_load_b128
            if (r >= rc.x && r < rc.y) {         // wave-uniform branch
                float z = zS[jb];
                for (int c = rc.z + lane; c < rc.w; c += 32) Hs[rr][c] = z;
            }
        }
    }
    __syncthreads();                    // T + Hs ready

    // --- WMMA column-upsample + fused loss ------------------------------
    const int hi16 = lane >> 4;
    const int mrow = lane & 15;
    float acc = 0.f;

    #pragma unroll
    for (int t = 0; t < 5; ++t) {
        const int j  = wv + 8 * t;              // 16-col output tile
        const int kc = j ? (4 * j - 2) : 0;

        // A frag: 16-bit A 16x32 layout (lanes 0-15: K0..7/K16..23,
        // lanes 16-31: K8..15/K24..31), M = lane&15
        v16h a;
        {
            const _Float16* trow = &T[mrow][0];
            int k0 = kc + 8 * hi16;
            #pragma unroll
            for (int i = 0; i < 8; ++i) {
                a[i]     = trow[k0 + i];
                a[8 + i] = trow[k0 + 16 + i];
            }
        }
        // B frag: 16-bit B 32x16 layout: N = lane&15, lanes 0-15 hold
        // K0..7/K16..23, lanes 16-31 hold K8..15/K24..31 (Bm stored col-major)
        v16h bf;
        {
            const _Float16* bcol = &Bm[16 * j + mrow][0];
            #pragma unroll
            for (int i = 0; i < 8; ++i) {
                bf[i]     = bcol[8 * hi16 + i];
                bf[8 + i] = bcol[8 * hi16 + 16 + i];
            }
        }

        v8f cz = {};
        v8f d = __builtin_amdgcn_wmma_f32_16x16x32_f16(
            false, a, false, bf, (short)0, cz, false, false);

        // D layout: lanes 0-15 -> M=v, lanes 16-31 -> M=v+8 ; N = lane&15
        const int col = 16 * j + mrow;
        #pragma unroll
        for (int v = 0; v < 8; ++v) {
            float e = d[v] - Hs[v + 8 * hi16][col];   // batched ds_load_b32
            acc += e * e;
        }
    }

    // --- deterministic block reduction ---
    redS[tid] = acc;
    __syncthreads();
    for (int s = 128; s > 0; s >>= 1) {
        if (tid < s) redS[tid] += redS[tid + s];
        __syncthreads();
    }
    if (tid == 0) partial[blockIdx.y * STRIPS + blockIdx.x] = redS[0];
}

__global__ __launch_bounds__(256) void cft_final(const float* __restrict__ partial,
                                                 float* __restrict__ out)
{
    __shared__ float r[256];
    float s = 0.f;
    for (int i = threadIdx.x; i < NBLK; i += 256) s += partial[i];
    r[threadIdx.x] = s;
    __syncthreads();
    for (int k = 128; k > 0; k >>= 1) {
        if (threadIdx.x < k) r[threadIdx.x] += r[threadIdx.x + k];
        __syncthreads();
    }
    if (threadIdx.x == 0)
        out[0] = r[0] * (1.0f / 13107200.0f);   // mean over 32*640*640
}

extern "C" void kernel_launch(void* const* d_in, const int* in_sizes, int n_in,
                              void* d_out, int out_size, void* d_ws, size_t ws_size,
                              hipStream_t stream) {
    const float* feat  = (const float*)d_in[0];   // [32,1,160,160] f32
    const float* boxes = (const float*)d_in[1];   // [32,128,5]     f32
    float* partials = (float*)d_ws;               // 1280 floats scratch

    dim3 grid(STRIPS, NBATCH);
    cft_main<<<grid, 256, 0, stream>>>(feat, boxes, partials);
    cft_final<<<1, 256, 0, stream>>>(partials, (float*)d_out);
}
